// TransformerDecoderLayer_5171140624713
// MI455X (gfx1250) — matrix operand order repaired and checked by reading, not compiled, not run on hardware
//
#include <hip/hip_runtime.h>

typedef __attribute__((ext_vector_type(16))) _Float16 v16h;
typedef __attribute__((ext_vector_type(8)))  float    v8f;

#define WMMA16(A, B, C) \
  __builtin_amdgcn_wmma_f32_16x16x32_f16(false, (A), false, (B), (short)0, (C), false, false)

constexpr int   Bc = 4, Nc = 2048, Dc = 512, HEc = 4, DHc = 64;
constexpr float LAMBDA_INIT = 0.8f;
constexpr float EPSc = 1e-5f;

// One 16-byte async copy, global -> LDS (GVS mode: saddr base + 32-bit voffset).
// Tracked by ASYNCcnt; drained with s_wait_asynccnt.
__device__ __forceinline__ void async_ld16(unsigned lds_addr, unsigned goff,
                                           const void* base)
{
    asm volatile("global_load_async_to_lds_b128 %0, %1, %2"
                 :: "v"(lds_addr), "v"(goff), "s"(base) : "memory");
}

// Low 32 bits of a generic pointer to __shared__ == LDS byte address.
__device__ __forceinline__ unsigned lds_off(const void* p)
{
    return (unsigned)(uintptr_t)p;
}

// ---------------------------------------------------------------------------
// Kernel A: pack K and V into head-major f16 planes so attention tiles are
// contiguous for the async DMA path (and no f32->f16 VALU in the hot loop).
//   k16[(b*HE+h)*2+s][n][64] ,  v16[b*HE+h][n][128]
// ---------------------------------------------------------------------------
__global__ __launch_bounds__(256)
void pack_kv_kernel(const float* __restrict__ k, const float* __restrict__ v,
                    _Float16* __restrict__ k16, _Float16* __restrict__ v16)
{
    int i4 = (blockIdx.x * 256 + threadIdx.x) * 4;          // 4 elems per thread
    if (i4 >= Bc * Nc * Dc) return;
    int d = i4 % Dc, n = (i4 / Dc) % Nc, b = i4 / (Dc * Nc);

    const float4 kv = *(const float4*)(k + i4);
    const float4 vv = *(const float4*)(v + i4);

    int h2 = d >> 6, h = h2 >> 1, s = h2 & 1, dh = d & 63;  // K: sub-head planes
    size_t ko = ((((size_t)(b * HEc + h) * 2 + s) * Nc) + n) * DHc + dh;
    k16[ko + 0] = (_Float16)kv.x; k16[ko + 1] = (_Float16)kv.y;
    k16[ko + 2] = (_Float16)kv.z; k16[ko + 3] = (_Float16)kv.w;

    int hv = d >> 7, dim = d & 127;                         // V: 128-wide heads
    size_t vo = (((size_t)(b * HEc + hv)) * Nc + n) * (2 * DHc) + dim;
    v16[vo + 0] = (_Float16)vv.x; v16[vo + 1] = (_Float16)vv.y;
    v16[vo + 2] = (_Float16)vv.z; v16[vo + 3] = (_Float16)vv.w;
}

// ---------------------------------------------------------------------------
// Kernel B: fused differential flash attention with double-buffered async
// K/V staging. grid (N/128, HE, B), 256 threads (8 wave32), 16 q-rows/wave.
// ---------------------------------------------------------------------------
__global__ __launch_bounds__(256)
void diff_attn_kernel(const float* __restrict__ q,
                      const _Float16* __restrict__ k16,
                      const _Float16* __restrict__ v16,
                      const float* __restrict__ lq1, const float* __restrict__ lk1,
                      const float* __restrict__ lq2, const float* __restrict__ lk2,
                      const float* __restrict__ norm_w,
                      _Float16* __restrict__ attn_out)
{
    // rows padded to multiples of 16B so async B128 destinations stay aligned
    __shared__ __align__(16) _Float16 sK[2][2][32][72];   // [buf][s][key][dh]
    __shared__ __align__(16) _Float16 sV[2][32][136];     // [buf][key][dim]
    __shared__ __align__(16) _Float16 sP[8][2][16][36];   // per-wave P staging
    __shared__ float s_lam;

    const int tid    = threadIdx.x;
    const int wave   = tid >> 5;
    const int lane   = tid & 31;
    const int half16 = lane >> 4;
    const int lm     = lane & 15;
    const int b      = blockIdx.z;
    const int h      = blockIdx.y;
    const int qrow0  = blockIdx.x * 128 + wave * 16;

    if (tid == 0) {
        float s1 = 0.f, s2 = 0.f;
        for (int i = 0; i < DHc; ++i) { s1 += lq1[i] * lk1[i]; s2 += lq2[i] * lk2[i]; }
        s_lam = __expf(s1) - __expf(s2) + LAMBDA_INIT;
    }

    // ---- async staging geometry (per-thread chunk assignment) ----
    const _Float16* kp0 = k16 + (((size_t)(b * HEc + h) * 2 + 0) * Nc) * DHc;
    const _Float16* kp1 = k16 + (((size_t)(b * HEc + h) * 2 + 1) * Nc) * DHc;
    const _Float16* vp  = v16 + ((size_t)(b * HEc + h) * Nc) * (2 * DHc);
    const int rowK = tid >> 3, chK = tid & 7;     // K tile: 32 rows x 8 x 16B
    const int rowV = tid >> 4, chV = tid & 15;    // V tile: 32 rows x 16 x 16B
    const unsigned sK_base = lds_off(&sK[0][0][0][0]);
    const unsigned sV_base = lds_off(&sV[0][0][0]);
    const unsigned ldsK_row = rowK * (72 * 2) + chK * 16;
    const unsigned ldsV_row = rowV * (136 * 2) + chV * 16;

    #define ISSUE_TILE(kb, buf)                                                    \
        do {                                                                       \
            async_ld16(sK_base + (unsigned)(((buf) * 2 + 0) * 32 * 144) + ldsK_row,\
                       (unsigned)(((kb) + rowK) * 128 + chK * 16), kp0);           \
            async_ld16(sK_base + (unsigned)(((buf) * 2 + 1) * 32 * 144) + ldsK_row,\
                       (unsigned)(((kb) + rowK) * 128 + chK * 16), kp1);           \
            async_ld16(sV_base + (unsigned)((buf) * 32 * 272) + ldsV_row,          \
                       (unsigned)(((kb) + rowV) * 256 + chV * 16), vp);            \
            async_ld16(sV_base + (unsigned)((buf) * 32 * 272 + 16 * 272) + ldsV_row,\
                       (unsigned)(((kb) + rowV + 16) * 256 + chV * 16), vp);       \
        } while (0)

    // ---- Q A-fragments [subhead][dh-half], scaled by DH^-0.5 ----
    v16h qa[2][2];
    #pragma unroll
    for (int s = 0; s < 2; ++s)
        #pragma unroll
        for (int j = 0; j < 2; ++j)
            #pragma unroll
            for (int vv = 0; vv < 8; ++vv) {
                int kk = ((vv >= 4) ? 16 : 0) + 8 * half16 + ((vv & 3) << 1);
                const float* qp = q + ((size_t)b * Nc + qrow0 + lm) * Dc
                                    + (2 * h + s) * DHc + j * 32 + kk;
                qa[s][j][2 * vv]     = (_Float16)(qp[0] * 0.125f);
                qa[s][j][2 * vv + 1] = (_Float16)(qp[1] * 0.125f);
            }

    v8f  acc[2][8];
    float rm[2][8], rl[2][8];
    #pragma unroll
    for (int s = 0; s < 2; ++s)
        #pragma unroll
        for (int t = 0; t < 8; ++t) {
            v8f z = {};
            acc[s][t] = z; rm[s][t] = -1e30f; rl[s][t] = 0.f;
        }
    float nw[8];
    #pragma unroll
    for (int t = 0; t < 8; ++t) nw[t] = norm_w[t * 16 + lm];

    constexpr int NT = Nc / 32;
    ISSUE_TILE(0, 0);                                   // prologue: tile 0 -> buf 0

    for (int kt = 0; kt < NT; ++kt) {
        const int cur = kt & 1;
        if (kt + 1 < NT) {
            ISSUE_TILE((kt + 1) * 32, cur ^ 1);         // pipeline next tile
            asm volatile("s_wait_asynccnt 0x4" ::: "memory");  // tile kt landed
        } else {
            asm volatile("s_wait_asynccnt 0x0" ::: "memory");
        }
        __syncthreads();                                // cross-wave visibility

        // ---- scores + online softmax per subhead ----
        #pragma unroll
        for (int s = 0; s < 2; ++s) {
            v8f sc[2];
            #pragma unroll
            for (int t = 0; t < 2; ++t) {
                v8f c = {};
                #pragma unroll
                for (int j = 0; j < 2; ++j) {
                    v16h bf;
                    #pragma unroll
                    for (int vv = 0; vv < 8; ++vv) {
                        int kk0 = ((vv >= 4) ? 16 : 0) + 8 * half16 + ((vv & 3) << 1);
                        bf[2 * vv]     = sK[cur][s][t * 16 + lm][j * 32 + kk0];
                        bf[2 * vv + 1] = sK[cur][s][t * 16 + lm][j * 32 + kk0 + 1];
                    }
                    c = WMMA16(qa[s][j], bf, c);
                }
                sc[t] = c;
            }
            #pragma unroll
            for (int r = 0; r < 8; ++r) {               // row = r + 8*half16
                float mx = fmaxf(sc[0][r], sc[1][r]);
                #pragma unroll
                for (int m = 1; m < 16; m <<= 1) mx = fmaxf(mx, __shfl_xor(mx, m, 32));
                float newm = fmaxf(rm[s][r], mx);
                float corr = __expf(rm[s][r] - newm);
                float p0 = __expf(sc[0][r] - newm);
                float p1 = __expf(sc[1][r] - newm);
                float rs = p0 + p1;
                #pragma unroll
                for (int m = 1; m < 16; m <<= 1) rs += __shfl_xor(rs, m, 32);
                rl[s][r] = rl[s][r] * corr + rs;
                rm[s][r] = newm;
                #pragma unroll
                for (int t = 0; t < 8; ++t) acc[s][t][r] *= corr;
                sP[wave][s][8 * half16 + r][lm]      = (_Float16)p0;
                sP[wave][s][8 * half16 + r][16 + lm] = (_Float16)p1;
            }
        }
        asm volatile("s_wait_dscnt 0x0" ::: "memory");  // per-wave LDS RAW drain

        // ---- P A-fragments, then P@V with shared V fragment ----
        v16h pa[2];
        #pragma unroll
        for (int s = 0; s < 2; ++s)
            #pragma unroll
            for (int vv = 0; vv < 8; ++vv) {
                int kk = ((vv >= 4) ? 16 : 0) + 8 * half16 + ((vv & 3) << 1);
                pa[s][2 * vv]     = sP[wave][s][lm][kk];
                pa[s][2 * vv + 1] = sP[wave][s][lm][kk + 1];
            }
        #pragma unroll
        for (int t = 0; t < 8; ++t) {
            v16h vf;
            #pragma unroll
            for (int vv = 0; vv < 8; ++vv) {
                int kk0 = ((vv >= 4) ? 16 : 0) + 8 * half16 + ((vv & 3) << 1);
                vf[2 * vv]     = sV[cur][kk0][t * 16 + lm];
                vf[2 * vv + 1] = sV[cur][kk0 + 1][t * 16 + lm];
            }
            acc[0][t] = WMMA16(pa[0], vf, acc[0][t]);
            acc[1][t] = WMMA16(pa[1], vf, acc[1][t]);
        }
        __syncthreads();                                // buf free for next issue
    }
    #undef ISSUE_TILE

    // ---- epilogue: diff, RMSNorm(128), affine, store f16 activations ----
    const float lam = s_lam;
    #pragma unroll
    for (int r = 0; r < 8; ++r) {
        float i0 = 1.0f / rl[0][r];
        float i1 = lam / rl[1][r];
        float ss = 0.f;
        #pragma unroll
        for (int t = 0; t < 8; ++t) {
            float dv = acc[0][t][r] * i0 - acc[1][t][r] * i1;
            ss += dv * dv;
        }
        #pragma unroll
        for (int m = 1; m < 16; m <<= 1) ss += __shfl_xor(ss, m, 32);
        float rms = rsqrtf(ss * (1.0f / 128.0f) + EPSc) * (1.0f - LAMBDA_INIT);
        int row = qrow0 + 8 * half16 + r;
        #pragma unroll
        for (int t = 0; t < 8; ++t) {
            float dv = acc[0][t][r] * i0 - acc[1][t][r] * i1;
            attn_out[((size_t)b * Nc + row) * Dc + h * 128 + t * 16 + lm] =
                (_Float16)(dv * rms * nw[t]);
        }
    }
}

// ---------------------------------------------------------------------------
// Kernel C: f32 -> f16 weight conversion.
// ---------------------------------------------------------------------------
__global__ __launch_bounds__(256)
void f32_to_f16_kernel(const float* __restrict__ src, _Float16* __restrict__ dst, int n)
{
    int i = blockIdx.x * blockDim.x + threadIdx.x;
    if (i < n) dst[i] = (_Float16)src[i];
}

// ---------------------------------------------------------------------------
// Kernel D: out = X(8192x512,f16) @ W^T(512x512,f16) + b, f32 result.
// 128x64 block tile, K-step 32, double-buffered async LDS staging.
// ---------------------------------------------------------------------------
__global__ __launch_bounds__(256)
void out_proj_kernel(const _Float16* __restrict__ x, const _Float16* __restrict__ wh,
                     const float* __restrict__ bias, float* __restrict__ out)
{
    __shared__ __align__(16) _Float16 sX[2][128][40];   // 80B rows (16B multiple)
    __shared__ __align__(16) _Float16 sW[2][64][40];
    const int tid = threadIdx.x, wave = tid >> 5, lane = tid & 31;
    const int half16 = lane >> 4, lm = lane & 15;
    const int row0 = blockIdx.x * 128;
    const int col0 = blockIdx.y * 64;

    const int rowX = tid >> 2, chX = tid & 3;           // X tile: 128 x 4 x 16B
    const int rowW = tid >> 2, chW = tid & 3;           // W tile:  64 x 4 x 16B
    const unsigned sX_base = lds_off(&sX[0][0][0]);
    const unsigned sW_base = lds_off(&sW[0][0][0]);

    #define ISSUE_GEMM(kb, buf)                                                     \
        do {                                                                        \
            async_ld16(sX_base + (unsigned)(((buf) * 128 + rowX) * 80 + chX * 16),  \
                       (unsigned)((row0 + rowX) * (Dc * 2) + (kb) * 2 + chX * 16), x); \
            async_ld16(sX_base + (unsigned)(((buf) * 128 + rowX + 64) * 80 + chX * 16), \
                       (unsigned)((row0 + rowX + 64) * (Dc * 2) + (kb) * 2 + chX * 16), x); \
            async_ld16(sW_base + (unsigned)(((buf) * 64 + rowW) * 80 + chW * 16),   \
                       (unsigned)((col0 + rowW) * (Dc * 2) + (kb) * 2 + chW * 16), wh); \
        } while (0)

    v8f acc[4];
    #pragma unroll
    for (int t = 0; t < 4; ++t) { v8f z = {}; acc[t] = z; }

    constexpr int KT = Dc / 32;
    ISSUE_GEMM(0, 0);

    for (int ki = 0; ki < KT; ++ki) {
        const int cur = ki & 1;
        if (ki + 1 < KT) {
            ISSUE_GEMM((ki + 1) * 32, cur ^ 1);
            asm volatile("s_wait_asynccnt 0x3" ::: "memory");
        } else {
            asm volatile("s_wait_asynccnt 0x0" ::: "memory");
        }
        __syncthreads();

        v16h af;
        #pragma unroll
        for (int vv = 0; vv < 8; ++vv) {
            int kk = ((vv >= 4) ? 16 : 0) + 8 * half16 + ((vv & 3) << 1);
            af[2 * vv]     = sX[cur][wave * 16 + lm][kk];
            af[2 * vv + 1] = sX[cur][wave * 16 + lm][kk + 1];
        }
        #pragma unroll
        for (int t = 0; t < 4; ++t) {
            v16h bf;
            #pragma unroll
            for (int vv = 0; vv < 8; ++vv) {
                int kk0 = ((vv >= 4) ? 16 : 0) + 8 * half16 + ((vv & 3) << 1);
                bf[2 * vv]     = sW[cur][t * 16 + lm][kk0];
                bf[2 * vv + 1] = sW[cur][t * 16 + lm][kk0 + 1];
            }
            acc[t] = WMMA16(af, bf, acc[t]);
        }
        __syncthreads();
    }
    #undef ISSUE_GEMM

    #pragma unroll
    for (int t = 0; t < 4; ++t) {
        int col = col0 + t * 16 + lm;
        float bv = bias[col];
        #pragma unroll
        for (int r = 0; r < 8; ++r) {
            int row = row0 + wave * 16 + 8 * half16 + r;
            out[(size_t)row * Dc + col] = acc[t][r] + bv;
        }
    }
}

// ---------------------------------------------------------------------------
extern "C" void kernel_launch(void* const* d_in, const int* in_sizes, int n_in,
                              void* d_out, int out_size, void* d_ws, size_t ws_size,
                              hipStream_t stream)
{
    const float* q      = (const float*)d_in[0];
    const float* k      = (const float*)d_in[1];
    const float* v      = (const float*)d_in[2];
    const float* lq1    = (const float*)d_in[3];
    const float* lk1    = (const float*)d_in[4];
    const float* lq2    = (const float*)d_in[5];
    const float* lk2    = (const float*)d_in[6];
    const float* norm_w = (const float*)d_in[7];
    const float* out_w  = (const float*)d_in[8];
    const float* out_b  = (const float*)d_in[9];
    float*       out    = (float*)d_out;

    // workspace layout (f16):  attn 8MB | W 0.5MB | K planes 8MB | V planes 8MB
    _Float16* attn = (_Float16*)d_ws;
    _Float16* wh   = attn + (size_t)Bc * Nc * Dc;
    _Float16* k16  = wh   + (size_t)Dc * Dc;
    _Float16* v16  = k16  + (size_t)Bc * Nc * Dc;

    pack_kv_kernel<<<(Bc * Nc * Dc / 4 + 255) / 256, 256, 0, stream>>>(k, v, k16, v16);
    f32_to_f16_kernel<<<(Dc * Dc + 255) / 256, 256, 0, stream>>>(out_w, wh, Dc * Dc);
    diff_attn_kernel<<<dim3(Nc / 128, HEc, Bc), 256, 0, stream>>>(
        q, k16, v16, lq1, lk1, lq2, lk2, norm_w, attn);
    out_proj_kernel<<<dim3((Bc * Nc) / 128, Dc / 64), 256, 0, stream>>>(
        attn, wh, out_b, out);
}
